// SelfAttentiveRotation_90366111907852
// MI455X (gfx1250) — matrix operand order repaired and checked by reading, not compiled, not run on hardware
//
#include <hip/hip_runtime.h>
#include <hip/hip_bf16.h>
#include <math.h>
#include <cstdint>

// ---- problem constants (match reference setup_inputs) ----
constexpr int B  = 1024;
constexpr int F  = 64;
constexpr int D  = 512;
constexpr int W  = 512;
constexpr int H  = 8;
constexpr int WH = W / H;   // 64 == F
constexpr float LN_EPS = 1e-5f;

typedef __attribute__((ext_vector_type(16))) __bf16 bf16x16;
typedef __attribute__((ext_vector_type(4)))  __bf16 bf16x4;
typedef __attribute__((ext_vector_type(8)))  float  f32x8;
typedef __attribute__((ext_vector_type(4)))  float  f32x4;

// ----------------------------------------------------------------------
// CDNA5 async copy: global -> LDS, 16 bytes per lane, tracked by ASYNCcnt.
// LDS destination address is the low 32 bits of the flat shared pointer
// (ISA: LDS aperture maps addr[31:0] to the LDS offset).
// ----------------------------------------------------------------------
__device__ __forceinline__ uint32_t lds_off(const void* p) {
  return (uint32_t)(uintptr_t)p;
}

__device__ __forceinline__ void async_ld_b128(uint32_t lds_byte_off, const void* gaddr) {
  asm volatile("global_load_async_to_lds_b128 %0, %1, off"
               :: "v"(lds_byte_off), "v"(gaddr)
               : "memory");
}

__device__ __forceinline__ void wait_async0() {
  asm volatile("s_wait_asynccnt 0x0" ::: "memory");
}
// wait until only the most recent group of 13 async copies is outstanding
__device__ __forceinline__ void wait_async13() {
  asm volatile("s_wait_asynccnt 0xd" ::: "memory");
}

// ----------------------------------------------------------------------
// WMMA fragment loaders (wave32, V_WMMA_F32_16X16X32_BF16 layouts)
//
// A (16x32, MxK), 16 bf16/lane:  lane L -> M = L&15, half = L>>4
//   e in [0,8):  K = half*8 + e          (VGPR 0..3)
//   e in [8,16): K = 16 + half*8 + (e-8) (VGPR 4..7)
// B (32x16, KxN), 16 bf16/lane:  lane L -> N = L&15, half = L>>4
//   e in [0,8):  K = half*8 + e; e in [8,16): K = 16 + half*8 + (e-8)
// C/D (16x16 f32, 8 VGPR): lane L -> N = L&15; reg r -> M = r + 8*(L>>4)
// ----------------------------------------------------------------------

template <typename T>
__device__ __forceinline__ bf16x16 ld_afrag(const T* __restrict__ src, long ld) {
  int lane = threadIdx.x & 31;
  int m = lane & 15, half = lane >> 4;
  const T* row = src + (long)m * ld;
  bf16x16 a;
#pragma unroll
  for (int e = 0; e < 8; ++e) {
    a[e]     = (__bf16)row[half * 8 + e];
    a[8 + e] = (__bf16)row[16 + half * 8 + e];
  }
  return a;
}

// generic-stride B fragment: element (kk,n) at src[kk*ldk + n*ldn]
template <typename T>
__device__ __forceinline__ bf16x16 ld_bfrag(const T* __restrict__ src, long ldk, long ldn) {
  int lane = threadIdx.x & 31;
  int n = lane & 15, half = lane >> 4;
  bf16x16 b;
#pragma unroll
  for (int e = 0; e < 16; ++e)
    b[e] = (__bf16)src[(long)((e < 8) ? (half * 8 + e) : (16 + half * 8 + (e - 8))) * ldk + (long)n * ldn];
  return b;
}

__device__ __forceinline__ f32x8 wmma_bf16(bf16x16 a, bf16x16 b, f32x8 c) {
  // 8 args: (neg_a, A, neg_b, B, c_mod, C, reuse_a, reuse_b)
  return __builtin_amdgcn_wmma_f32_16x16x32_bf16(false, a, false, b, (short)0, c, false, false);
}

// ----------------------------------------------------------------------
// Kernel 1: fused q/k/v projections.
// t[b,f,w] = sum_d theta[b,f,d] * P[f,d,w]   for P in {Q,K,V}
// 128 threads (4 waves), block tile = 16 batch rows x 64 w cols.
// Software pipeline: 13 async b128 copies stage the NEXT K-step's tiles
// (theta 16x32 + 3x P 32x64) into the ping-pong LDS buffer while the
// CURRENT step's 12 WMMAs run; s_wait_asynccnt 13 retires only the older
// group. A tile is read by all 4 waves; each wave reads its own 16-col B slice.
// ----------------------------------------------------------------------
__global__ void __launch_bounds__(128)
proj_qkv_kernel(const float* __restrict__ theta,
                const float* __restrict__ Qw,
                const float* __restrict__ Kw,
                const float* __restrict__ Vw,
                float* __restrict__ q_out,
                float* __restrict__ k_out,
                float* __restrict__ v_out) {
  constexpr int NTB   = W / 64;              // 8 n-blocks of 64 cols
  constexpr int STEPS = D / 32;              // 16 K-steps
  const int mt = blockIdx.x / NTB;           // batch tile (16 rows)
  const int nb = blockIdx.x % NTB;           // w block (64 cols)
  const int f  = blockIdx.y;
  const int tid = threadIdx.x;
  const int wid = tid >> 5;
  const int lane = tid & 31, n = lane & 15, half = lane >> 4;
  const int n0 = nb * 64;

  __shared__ __align__(16) float s_a[2][16 * 32];        // theta tiles (ping-pong)
  __shared__ __align__(16) float s_b[2][3][32 * 64];     // Q/K/V tiles (ping-pong)

  const float* Abase = theta + (long)(mt * 16) * F * D + (long)f * D; // + m*F*D + d
  const float* const Pf[3] = { Qw + (long)f * D * W,
                               Kw + (long)f * D * W,
                               Vw + (long)f * D * W };

  // staging maps (per thread, 16B each)
  const int a_row = tid >> 3;              // 0..15
  const int a_c4  = (tid & 7) * 4;         // float col in [0,32)
  const int b_row = tid >> 4;              // 0..7
  const int b_c4  = (tid & 15) * 4;        // float col in [0,64)

  auto stage = [&](int buf, int k0) {
    async_ld_b128(lds_off(&s_a[buf][a_row * 32 + a_c4]),
                  Abase + (long)a_row * (F * D) + k0 + a_c4);
#pragma unroll
    for (int p = 0; p < 3; ++p) {
#pragma unroll
      for (int it = 0; it < 4; ++it) {
        int kk = it * 8 + b_row;
        async_ld_b128(lds_off(&s_b[buf][p][kk * 64 + b_c4]),
                      Pf[p] + (long)(k0 + kk) * W + n0 + b_c4);
      }
    }
  };

  f32x8 acc[3] = {f32x8{}, f32x8{}, f32x8{}};

  stage(0, 0);                                  // prologue
  for (int step = 0; step < STEPS; ++step) {
    const int cur = step & 1;
    if (step + 1 < STEPS) {
      stage(cur ^ 1, (step + 1) * 32);          // overlap DMA with compute
      wait_async13();                           // retire only the older group
    } else {
      wait_async0();
    }
    __syncthreads();                            // cur buffer visible to all waves

    bf16x16 a = ld_afrag(s_a[cur], 32);
#pragma unroll
    for (int p = 0; p < 3; ++p) {
      bf16x16 bf = ld_bfrag(&s_b[cur][p][wid * 16], 64, 1);
      acc[p] = wmma_bf16(a, bf, acc[p]);
    }
    __syncthreads();                            // all reads of cur done before overwrite
  }

#pragma unroll
  for (int r = 0; r < 8; ++r) {
    int m = r + 8 * half;
    long off = (long)(mt * 16 + m) * F * W + (long)f * W + n0 + wid * 16 + n;
    q_out[off] = acc[0][r];
    k_out[off] = acc[1][r];
    v_out[off] = acc[2][r];
  }
}

// ----------------------------------------------------------------------
// Kernel 2: rotation attention per (b,h).
// S[f,g]   = sigmoid( sum_w cos(q[f,w])*wt[w]*cos(k[g,w])
//                   + sum_w sin(q[f,w])*wt[w]*sin(k[g,w]) )
// out[b, f, h*64+g] = sum_w S[f,w] * v[g,w]
// 128 threads (4 waves). v tile staged by 8 async b128 copies that overlap
// the f32 transcendental work; q/k read as float4, results stored as packed
// bf16x4 (8B DS stores).
// ----------------------------------------------------------------------
__global__ void __launch_bounds__(128)
attn_kernel(const float* __restrict__ q,
            const float* __restrict__ k,
            const float* __restrict__ v,
            const float* __restrict__ weight,
            float* __restrict__ out) {
  const int bh = blockIdx.x;
  const int b = bh >> 3;
  const int h = bh & 7;
  const int tid = threadIdx.x;

  __shared__ __align__(16) __bf16 s_cq[64 * 64];
  __shared__ __align__(16) __bf16 s_sq[64 * 64];
  __shared__ __align__(16) __bf16 s_ck[64 * 64];
  __shared__ __align__(16) __bf16 s_sk[64 * 64];
  __shared__ __align__(16) float  s_vf[64 * 64];
  __shared__ __align__(16) __bf16 s_att[64 * 64];

  const long base = (long)b * F * W + (long)h * WH;  // + f*W + w

  // stage v tile asynchronously (64 rows x 64 f32, 16B per lane per op)
  {
    const float* vbase = v + base;
    const int fr0 = tid >> 4;            // 0..7
    const int c4  = (tid & 15) * 4;      // float col in [0,64)
#pragma unroll
    for (int it = 0; it < 8; ++it) {
      int fr = it * 8 + fr0;
      async_ld_b128(lds_off(&s_vf[fr * 64 + c4]), vbase + (long)fr * W + c4);
    }
  }

  // cos/sin operands: float4 loads, packed bf16x4 stores (overlaps async DMA)
#pragma unroll
  for (int it = 0; it < 8; ++it) {
    int idx4 = tid + 128 * it;           // 1024 float4 chunks total
    int fi = idx4 >> 4;                  // row (0..63)
    int c  = (idx4 & 15) * 4;            // col (0..60)
    f32x4 qv = *(const f32x4*)(q + base + (long)fi * W + c);
    f32x4 kv = *(const f32x4*)(k + base + (long)fi * W + c);
    f32x4 wt = *(const f32x4*)(weight + c);
    bf16x4 cq, sq, ck4, sk4;
#pragma unroll
    for (int e = 0; e < 4; ++e) {
      cq[e]  = (__bf16)(cosf(qv[e]) * wt[e]);
      sq[e]  = (__bf16)(sinf(qv[e]) * wt[e]);
      ck4[e] = (__bf16)cosf(kv[e]);
      sk4[e] = (__bf16)sinf(kv[e]);
    }
    *(bf16x4*)&s_cq[fi * 64 + c] = cq;
    *(bf16x4*)&s_sq[fi * 64 + c] = sq;
    *(bf16x4*)&s_ck[fi * 64 + c] = ck4;
    *(bf16x4*)&s_sk[fi * 64 + c] = sk4;
  }
  wait_async0();
  __syncthreads();

  const int wid = tid >> 5;
  const int lane = tid & 31, n = lane & 15, half = lane >> 4;

  // phase 2: logits + sigmoid. wave `wid` owns row-tile mt = wid (f rows).
#pragma unroll
  for (int ntile = 0; ntile < 4; ++ntile) {
    f32x8 acc = {};
#pragma unroll
    for (int k0 = 0; k0 < 64; k0 += 32) {
      bf16x16 acq = ld_afrag(s_cq + wid * 16 * 64 + k0, 64);
      bf16x16 bck = ld_bfrag(s_ck + ntile * 16 * 64 + k0, (long)1, (long)64);  // k^T
      bf16x16 asq = ld_afrag(s_sq + wid * 16 * 64 + k0, 64);
      bf16x16 bsk = ld_bfrag(s_sk + ntile * 16 * 64 + k0, (long)1, (long)64);  // k^T
      acc = wmma_bf16(acq, bck, acc);
      acc = wmma_bf16(asq, bsk, acc);
    }
#pragma unroll
    for (int r = 0; r < 8; ++r) {
      float sg = 1.0f / (1.0f + expf(-acc[r]));
      s_att[(wid * 16 + r + 8 * half) * 64 + ntile * 16 + n] = (__bf16)sg;
    }
  }
  __syncthreads();

  // phase 3: out[f,g] = sum_w att[f,w] * v[g,w]
#pragma unroll
  for (int ntile = 0; ntile < 4; ++ntile) {
    f32x8 acc = {};
#pragma unroll
    for (int k0 = 0; k0 < 64; k0 += 32) {
      bf16x16 aa = ld_afrag(s_att + wid * 16 * 64 + k0, 64);
      bf16x16 bv = ld_bfrag(s_vf + ntile * 16 * 64 + k0, (long)1, (long)64);   // v^T
      acc = wmma_bf16(aa, bv, acc);
    }
#pragma unroll
    for (int r = 0; r < 8; ++r) {
      int fidx = wid * 16 + r + 8 * half;
      int g = ntile * 16 + n;
      out[(long)b * F * W + (long)fidx * W + h * WH + g] = acc[r];
    }
  }
}

// ----------------------------------------------------------------------
// Kernel 3: x = x_att + theta @ W_lin^T, then LayerNorm over W.
// 256 threads (8 waves) own 16 full rows (n = b*F+f flattened).
// Deterministic LDS reduction (no float atomics -> bit-stable replay).
// ----------------------------------------------------------------------
__global__ void __launch_bounds__(256)
linear_ln_kernel(const float* __restrict__ theta,
                 const float* __restrict__ wlin,
                 const float* __restrict__ gamma,
                 const float* __restrict__ beta,
                 float* __restrict__ out) {   // in: x_att, out: final
  const int row0 = blockIdx.x * 16;          // rows over N = B*F
  const int tid = threadIdx.x;
  const int wid = tid >> 5;
  const int lane = tid & 31, n = lane & 15, half = lane >> 4;

  __shared__ float s_ps [16][128];
  __shared__ float s_pss[16][128];
  __shared__ float s_mean[16], s_rstd[16];

  const float* Arow = theta + (long)row0 * D;

  f32x8 acc[4] = {f32x8{}, f32x8{}, f32x8{}, f32x8{}};
  for (int k0 = 0; k0 < D; k0 += 32) {
    if (k0 + 32 < D) __builtin_prefetch(Arow + k0 + 32, 0, 0);  // global_prefetch_b8
    bf16x16 a = ld_afrag(Arow + k0, D);
#pragma unroll
    for (int j = 0; j < 4; ++j) {
      int w0 = (wid * 4 + j) * 16;
      bf16x16 bw = ld_bfrag(wlin + (long)w0 * D + k0, (long)1, (long)D);  // W_lin^T
      acc[j] = wmma_bf16(a, bw, acc[j]);
    }
  }

  // residual add + deterministic partial sums
  float vals[4][8];
  const int widx = wid * 16 + n;
#pragma unroll
  for (int r = 0; r < 8; ++r) {
    int m = r + 8 * half;
    float ps = 0.f, pss = 0.f;
#pragma unroll
    for (int j = 0; j < 4; ++j) {
      int w = (wid * 4 + j) * 16 + n;
      float x = acc[j][r] + out[(long)(row0 + m) * W + w];
      vals[j][r] = x;
      ps += x;
      pss += x * x;
    }
    s_ps [m][widx] = ps;
    s_pss[m][widx] = pss;
  }
  __syncthreads();

  if (tid < 16) {
    float s = 0.f, ss = 0.f;
    for (int i = 0; i < 128; ++i) { s += s_ps[tid][i]; ss += s_pss[tid][i]; }
    float mean = s * (1.0f / W);
    float var  = ss * (1.0f / W) - mean * mean;
    s_mean[tid] = mean;
    s_rstd[tid] = rsqrtf(var + LN_EPS);
  }
  __syncthreads();

#pragma unroll
  for (int r = 0; r < 8; ++r) {
    int m = r + 8 * half;
    float mean = s_mean[m], rstd = s_rstd[m];
#pragma unroll
    for (int j = 0; j < 4; ++j) {
      int w = (wid * 4 + j) * 16 + n;
      out[(long)(row0 + m) * W + w] = (vals[j][r] - mean) * rstd * gamma[w] + beta[w];
    }
  }
}

// ----------------------------------------------------------------------
extern "C" void kernel_launch(void* const* d_in, const int* in_sizes, int n_in,
                              void* d_out, int out_size, void* d_ws, size_t ws_size,
                              hipStream_t stream) {
  const float* theta  = (const float*)d_in[0];
  const float* Qw     = (const float*)d_in[1];
  const float* Kw     = (const float*)d_in[2];
  const float* Vw     = (const float*)d_in[3];
  const float* Wl     = (const float*)d_in[4];
  const float* weight = (const float*)d_in[5];
  const float* gamma  = (const float*)d_in[6];
  const float* beta   = (const float*)d_in[7];
  float* out = (float*)d_out;

  const size_t BFW = (size_t)B * F * W;
  float* qb = (float*)d_ws;
  float* kb = qb + BFW;
  float* vb = kb + BFW;

  // 1) fused projections: q,k,v = theta @ {Q,K,V} per f
  dim3 g1((B / 16) * (W / 64), F, 1);
  proj_qkv_kernel<<<g1, 128, 0, stream>>>(theta, Qw, Kw, Vw, qb, kb, vb);

  // 2) rotation attention per (b,h) -> writes attention output into d_out
  attn_kernel<<<B * H, 128, 0, stream>>>(qb, kb, vb, weight, out);

  // 3) residual linear + layernorm (reads+overwrites d_out)
  linear_ln_kernel<<<(B * F) / 16, 256, 0, stream>>>(theta, Wl, gamma, beta, out);
}
